// MySupConLoss_49890340110909
// MI455X (gfx1250) — matrix operand order repaired and checked by reading, not compiled, not run on hardware
//
#include <hip/hip_runtime.h>
#include <hip/hip_bf16.h>
#include <math.h>

typedef _Float16 half_t;
typedef __attribute__((ext_vector_type(4)))  _Float16 v4h;
typedef __attribute__((ext_vector_type(8)))  _Float16 v8h;
typedef __attribute__((ext_vector_type(16))) _Float16 v16h;
typedef __attribute__((ext_vector_type(8)))  float    v8f;

#define NROWS 8192
#define NTILE (NROWS / 16)          /* 512 column tiles */
#define DIM   128
#define INV_T 14.2857142857142857f  /* 1/0.07 */

// ---------------- Kernel 0: zero the global loss accumulator ----------------
__global__ void supcon_init(float* __restrict__ accum) {
  if (threadIdx.x == 0) *accum = 0.0f;
}

// ------- Kernel 1: L2-normalize rows (f32 in), write f16 features -----------
__global__ void supcon_normalize(const float* __restrict__ f,
                                 half_t* __restrict__ fnh) {
  const int wave = threadIdx.x >> 5;
  const int lane = threadIdx.x & 31;
  const int row  = blockIdx.x * 8 + wave;   // blockDim.x == 256 -> 8 waves
  const float4 v = ((const float4*)(f + (size_t)row * DIM))[lane];
  float ss = v.x * v.x + v.y * v.y + v.z * v.z + v.w * v.w;
#pragma unroll
  for (int m = 1; m < 32; m <<= 1) ss += __shfl_xor(ss, m, 32);
  const float scale = 1.0f / fmaxf(sqrtf(ss), 1e-8f);
  v4h o;
  o[0] = (half_t)(v.x * scale);
  o[1] = (half_t)(v.y * scale);
  o[2] = (half_t)(v.z * scale);
  o[3] = (half_t)(v.w * scale);
  *(v4h*)(fnh + (size_t)row * DIM + lane * 4) = o;
}

// ---------------- helpers for the fused main kernel ----------------
__device__ __forceinline__ void load_b(const half_t* __restrict__ fnh, int jt,
                                       int l15, int hi16, v16h b[4]) {
  // B (32x16 f16): lane holds column n=l15; lanes 0-15 carry K0..15,
  // lanes 16-31 carry K16..31 -> one contiguous 32B load per lane per k.
  const half_t* pb = fnh + (size_t)(jt * 16 + l15) * DIM + hi16 * 16;
  b[0] = *(const v16h*)(pb);
  b[1] = *(const v16h*)(pb + 32);
  b[2] = *(const v16h*)(pb + 64);
  b[3] = *(const v16h*)(pb + 96);
}

__device__ __forceinline__ void load_m(const int* __restrict__ posm,
                                       const int* __restrict__ negm,
                                       size_t iBase, int j,
                                       int pm[8], int nm[8]) {
  // Nontemporal: one-shot 512MB stream must not evict L2-resident features.
  const int* pp = posm + iBase * NROWS + j;
  const int* np = negm + iBase * NROWS + j;
#pragma unroll
  for (int r = 0; r < 8; ++r) {
    pm[r] = __builtin_nontemporal_load(pp + (size_t)r * NROWS);
    nm[r] = __builtin_nontemporal_load(np + (size_t)r * NROWS);
  }
}

__device__ __forceinline__ void compute_tile(const v16h a[4], const v16h b[4],
                                             const int pm[8], const int nm[8],
                                             int i0, int j,
                                             float posAcc[8], float negAcc[8]) {
  v8f c = {};
#pragma unroll
  for (int k = 0; k < 4; ++k)
    c = __builtin_amdgcn_wmma_f32_16x16x32_f16(false, a[k], false, b[k],
                                               (short)0, c, false, false);
#pragma unroll
  for (int r = 0; r < 8; ++r) {
    float e = __expf(c[r] * INV_T);
    if (i0 + r == j) e = 0.0f;  // remove self-contrast (diagonal)
    posAcc[r] += e * (float)pm[r];
    negAcc[r] += e * (float)nm[r];
  }
}

// ------- Kernel 2: fused Gram(WMMA f16) -> exp -> masked row reduction ------
// grid = 512 blocks (16-row stripes), 256 threads = 8 waves.
// Each wave handles column tiles jt = wave, wave+8, ... (64 tiles), software
// pipelined with double buffers so each buffer's loads (8x b128 + 16x b32 NT)
// are covered by the other buffer's full compute phase.
__global__ void __launch_bounds__(256)
__attribute__((amdgpu_waves_per_eu(4)))
supcon_main(const half_t* __restrict__ fnh,
            const int*    __restrict__ posm,
            const int*    __restrict__ negm,
            float*        __restrict__ accum) {
  __shared__ float s_pos[16];
  __shared__ float s_neg[16];
  if (threadIdx.x < 16) { s_pos[threadIdx.x] = 0.0f; s_neg[threadIdx.x] = 0.0f; }
  __syncthreads();

  const int wave = threadIdx.x >> 5;
  const int lane = threadIdx.x & 31;
  const int hi16 = lane >> 4;   // 0: lanes 0-15, 1: lanes 16-31
  const int l15  = lane & 15;
  const int rowBase = blockIdx.x * 16;

  // Preload the 4 A fragments (16x32 f16, K = 0..127) once per wave.
  // Layout: lanes 0-15 hold row M=lane, K0..15;
  //         lanes 16-31 hold row M=lane&15, K16..31 (matches B).
  v16h a[4];
#pragma unroll
  for (int k = 0; k < 4; ++k) {
    const half_t* pa = fnh + (size_t)(rowBase + l15) * DIM + k * 32 + hi16 * 16;
    v8h lo = *(const v8h*)(pa);        // K offsets +0..7   (VGPR 0-3)
    v8h hi = *(const v8h*)(pa + 8);    // K offsets +8..15  (VGPR 4-7)
    a[k] = __builtin_shufflevector(lo, hi, 0, 1, 2, 3, 4, 5, 6, 7,
                                           8, 9, 10, 11, 12, 13, 14, 15);
  }

  float posAcc[8] = {0, 0, 0, 0, 0, 0, 0, 0};
  float negAcc[8] = {0, 0, 0, 0, 0, 0, 0, 0};
  const int    i0    = rowBase + 8 * hi16;       // row of c[r] is i0+r
  const size_t iBase = (size_t)i0;

  // ---- software pipeline: prologue fills buffer 0 ----
  v16h b0[4], b1[4];
  int  pm0[8], nm0[8], pm1[8], nm1[8];
  load_b(fnh, wave, l15, hi16, b0);
  load_m(posm, negm, iBase, wave * 16 + l15, pm0, nm0);

  for (int jt = wave; jt < NTILE; jt += 16) {
    const int jtB = jt + 8;                              // always < NTILE
    load_b(fnh, jtB, l15, hi16, b1);                     // prefetch buf 1
    load_m(posm, negm, iBase, jtB * 16 + l15, pm1, nm1);

    compute_tile(a, b0, pm0, nm0, i0, jt * 16 + l15, posAcc, negAcc);

    const int jtC = (jt + 16 < NTILE) ? jt + 16 : jt;    // clamped (uniform)
    load_b(fnh, jtC, l15, hi16, b0);                     // prefetch buf 0
    load_m(posm, negm, iBase, jtC * 16 + l15, pm0, nm0);

    compute_tile(a, b1, pm1, nm1, i0, jtB * 16 + l15, posAcc, negAcc);
  }

  // Row sums: reduce over the 16 lanes (column index) of each half-wave.
#pragma unroll
  for (int r = 0; r < 8; ++r) {
#pragma unroll
    for (int m = 1; m < 16; m <<= 1) {
      posAcc[r] += __shfl_xor(posAcc[r], m, 32);
      negAcc[r] += __shfl_xor(negAcc[r], m, 32);
    }
  }
  if (l15 == 0) {
#pragma unroll
    for (int r = 0; r < 8; ++r) {
      atomicAdd(&s_pos[r + 8 * hi16], posAcc[r]);  // ds_add_f32
      atomicAdd(&s_neg[r + 8 * hi16], negAcc[r]);
    }
  }
  __syncthreads();

  if (threadIdx.x < 16) {
    const float p = s_pos[threadIdx.x];
    const float n = s_neg[threadIdx.x];
    float t = __logf(p / (p + n));
#pragma unroll
    for (int m = 1; m < 16; m <<= 1) t += __shfl_xor(t, m, 32);
    if (threadIdx.x == 0) atomicAdd(accum, t);  // global_atomic_add_f32
  }
}

// ---------------- Kernel 3: finalize loss = -sum/N ----------------
__global__ void supcon_final(const float* __restrict__ accum,
                             float* __restrict__ out) {
  if (threadIdx.x == 0) out[0] = -(*accum) / (float)NROWS;
}

extern "C" void kernel_launch(void* const* d_in, const int* in_sizes, int n_in,
                              void* d_out, int out_size, void* d_ws,
                              size_t ws_size, hipStream_t stream) {
  const float* features = (const float*)d_in[0];
  const int*   posm     = (const int*)d_in[1];
  const int*   negm     = (const int*)d_in[2];
  float*       out      = (float*)d_out;

  // Workspace layout: [0..3] loss accumulator, [256..] f16 normalized
  // features (8192*128*2 = 2 MB).
  float*  accum = (float*)d_ws;
  half_t* fnh   = (half_t*)((char*)d_ws + 256);

  supcon_init<<<1, 32, 0, stream>>>(accum);
  supcon_normalize<<<NROWS / 8, 256, 0, stream>>>(features, fnh);
  supcon_main<<<NROWS / 16, 256, 0, stream>>>(fnh, posm, negm, accum);
  supcon_final<<<1, 32, 0, stream>>>(accum, out);
}
